// GeometryEncoding_v5_16999480557591
// MI455X (gfx1250) — matrix operand order repaired and checked by reading, not compiled, not run on hardware
//
#include <hip/hip_runtime.h>
#include <hip/hip_bf16.h>
#include <math.h>

// ---------------------------------------------------------------------------
// GeometryEncoding for MI455X (gfx1250, wave32, WMMA).
// All 64x64 / 64x128 1x1-convs run as v_wmma_f32_16x16x32_f16 GEMMs:
//   D[16x16] = A[16x32](weights,f16) x B[32x16](activations,f16) + C(f32)
// Weights are preloaded into registers (weight-stationary; __launch_bounds__
// (256,1) so nothing spills), activations stream as one contiguous 32B load
// per lane per K-step. BN+ReLU on the B operand is packed-f16 math.
// The polar bilinear scatter->normalize->gather is local to each point n,
// so it is done per-wave in LDS with no atomics (in place on P_raw).
// ---------------------------------------------------------------------------

typedef __attribute__((ext_vector_type(16))) _Float16 v16h;
typedef __attribute__((ext_vector_type(8)))  _Float16 v8h;
typedef __attribute__((ext_vector_type(8)))  float    v8f;

#define BDIM   8
#define NDIM   4096
#define KDIM   16
#define CDIM   64
#define POS    (BDIM*NDIM*KDIM)     // 524288 positions
#define NSTRIP (POS/16)             // 32768 WMMA column strips (one per (b,n))
#define NPTS   (BDIM*NDIM)          // 32768 points
#define EPS    1e-6f
#define BNEPS  1e-5f
#define PI_F   3.14159265358979323846f

// ---- workspace layout (bytes) ----
#define SZ_ACT   ((size_t)POS*CDIM*2)          // 64 MB per f16 activation buffer
#define OFF_HC   ((size_t)0)                   // h_c  -> reused as FUSED
#define OFF_HP   (SZ_ACT)                      // h_p  -> reused as H_G
#define OFF_CRAW (2*SZ_ACT)
#define OFF_PRAW (3*SZ_ACT)                    // P_raw -> P_aligned (in place)
#define OFF_IDX  (4*SZ_ACT)                    // packed bins, u32 per position
#define OFF_WTS  (OFF_IDX + (size_t)POS*4)     // 4 bilinear weights per position
#define OFF_WEI  (OFF_WTS + (size_t)POS*16)
#define OFF_WC2  (OFF_WEI)                     // 64x64  f16
#define OFF_WP2  (OFF_WEI +  8192)             // 64x64  f16
#define OFF_WG1  (OFF_WEI + 16384)             // 64x64  f16
#define OFF_WG2  (OFF_WEI + 24576)             // 64x64  f16
#define OFF_WFU  (OFF_WEI + 32768)             // 64x128 f16  [ac_w | gamma*ap_w]
#define OFF_BFU  (OFF_WEI + 49152)             // 64 f32
#define OFF_STAT (OFF_WEI + 49408)             // 3 x (sum[64], sumsq[64]) f32
#define OFF_COEF (OFF_STAT + 1536)             // 3 x (a[64], b[64]) f32
#define OFF_COEFH (OFF_COEF + 1536)            // 3 x (a[64], b[64]) f16

// ---------------------------------------------------------------------------
// K0: weight conversion to f16, gamma folding, stats zeroing
// ---------------------------------------------------------------------------
__global__ void k_prep(const float* cw2, const float* pw2, const float* gw1,
                       const float* gw2, const float* acw, const float* apw,
                       const float* acb, const float* apb, const float* gammap,
                       _Float16* WC2, _Float16* WP2, _Float16* WG1, _Float16* WG2,
                       _Float16* WFU, float* BFU, float* STATS) {
  const int t = threadIdx.x;
  const float gm = gammap[0];
  for (int i = t; i < 4096; i += 256) {
    WC2[i] = (_Float16)cw2[i]; WP2[i] = (_Float16)pw2[i];
    WG1[i] = (_Float16)gw1[i]; WG2[i] = (_Float16)gw2[i];
  }
  for (int i = t; i < 8192; i += 256) {
    int c = i >> 7, j = i & 127;
    WFU[i] = (_Float16)((j < 64) ? acw[c*64 + j] : gm * apw[c*64 + (j-64)]);
  }
  for (int i = t; i < 64;  i += 256) BFU[i]   = acb[i] + gm * apb[i];
  for (int i = t; i < 384; i += 256) STATS[i] = 0.f;
}

// ---------------------------------------------------------------------------
// K1: geometry + first convs (4->64, 5->64) + bin indices/weights
//     one thread per (b,n,k); k = lane&15 so r_mean is a 16-lane shuffle
// ---------------------------------------------------------------------------
__global__ void k_geom(const float* __restrict__ nx, const float* __restrict__ cx,
                       const float* __restrict__ cw1, const float* __restrict__ cb1,
                       const float* __restrict__ pw1, const float* __restrict__ pb1,
                       _Float16* __restrict__ hc, _Float16* __restrict__ hp,
                       unsigned* __restrict__ idxp, float* __restrict__ wts) {
  const int p = blockIdx.x * blockDim.x + threadIdx.x;
  if (p >= POS) return;
  const int k = p & 15;
  const int n = (p >> 4) & (NDIM - 1);
  const int b = p >> 16;                      // N*K = 65536 positions per batch

  const float dx = nx[((b*3+0)*NDIM + n)*KDIM + k] - cx[(b*3+0)*NDIM + n];
  const float dy = nx[((b*3+1)*NDIM + n)*KDIM + k] - cx[(b*3+1)*NDIM + n];
  const float dz = nx[((b*3+2)*NDIM + n)*KDIM + k] - cx[(b*3+2)*NDIM + n];

  const float r   = fmaxf(sqrtf(dx*dx + dy*dy + dz*dz), EPS);
  const float rho = fmaxf(sqrtf(dx*dx + dy*dy), EPS);
  const float th  = atan2f(dy, dx);
  const float ph  = atan2f(dz, rho);

  float rm = r;                               // mean(r) over K=16 (lanes 0..15 group)
  for (int off = 1; off < 16; off <<= 1) rm += __shfl_xor(rm, off, 32);
  rm *= (1.f/16.f);
  const float rnorm = fminf(r / (rm + EPS), 3.f) * (1.f/3.f);

  // bilinear bin indices / weights (R=2, A=12)
  const float rreal = rnorm * (2.f - 1e-6f);
  const float areal = (th + PI_F) * (1.f/(2.f*PI_F)) * (12.f - 1e-6f);
  const float rf = floorf(rreal), af = floorf(areal);
  const float rw1 = rreal - rf,  aw1 = areal - af;
  const float rw0 = 1.f - rw1,   aw0 = 1.f - aw1;
  int ri0 = min(max((int)rf, 0), 1),  ri1 = min(max((int)rf + 1, 0), 1);
  int ai0 = min(max((int)af, 0), 11), ai1 = min(max((int)af + 1, 0), 11);
  idxp[p] = (unsigned)(ri0 | (ri1 << 1) | (ai0 << 2) | (ai1 << 6));
  wts[(size_t)p*4 + 0] = rw0*aw0;  wts[(size_t)p*4 + 1] = rw0*aw1;
  wts[(size_t)p*4 + 2] = rw1*aw0;  wts[(size_t)p*4 + 3] = rw1*aw1;

  float st, ct, sp, cp;
  __sincosf(th, &st, &ct);
  __sincosf(ph, &sp, &cp);

  _Float16* hcw = hc + (size_t)p * CDIM;
  _Float16* hpw = hp + (size_t)p * CDIM;
  #pragma unroll 4
  for (int c = 0; c < CDIM; ++c) {
    float hv = cb1[c] + cw1[c*4+0]*dx + cw1[c*4+1]*dy + cw1[c*4+2]*dz + cw1[c*4+3]*r;
    hcw[c] = (_Float16)hv;
    float pv = pb1[c] + pw1[c*5+0]*rnorm + pw1[c*5+1]*st + pw1[c*5+2]*ct
                      + pw1[c*5+3]*sp    + pw1[c*5+4]*cp;
    hpw[c] = (_Float16)pv;
  }
}

// ---------------------------------------------------------------------------
// K2: per-channel sum/sumsq over all positions (for BatchNorm batch stats)
// ---------------------------------------------------------------------------
__global__ void k_stats(const _Float16* __restrict__ h, float* __restrict__ stats) {
  __shared__ float ssum[64], ssq[64];
  const int t = threadIdx.x, c = t & 63, g = t >> 6;
  if (t < 64) { ssum[t] = 0.f; ssq[t] = 0.f; }
  __syncthreads();
  float s = 0.f, sq = 0.f;
  for (size_t p = (size_t)blockIdx.x*4 + g; p < POS; p += (size_t)gridDim.x*4) {
    float v = (float)h[p*CDIM + c];
    s += v; sq += v*v;
  }
  atomicAdd(&ssum[c], s); atomicAdd(&ssq[c], sq);
  __syncthreads();
  if (t < 64) { atomicAdd(&stats[t], ssum[t]); atomicAdd(&stats[64+t], ssq[t]); }
}

// finalize BN: write f32 coefficients and packed-f16 copies for the GEMM path
__global__ void k_finalize(const float* __restrict__ stats, const float* __restrict__ g,
                           const float* __restrict__ be, float* __restrict__ coef,
                           _Float16* __restrict__ coefh) {
  const int c = threadIdx.x;
  if (c < 64) {
    const float inv = 1.f / (float)POS;
    float mean = stats[c] * inv;
    float var  = stats[64+c] * inv - mean*mean;     // biased variance
    float a    = g[c] * rsqrtf(var + BNEPS);
    float b    = be[c] - mean * a;
    coef[c]       = a;  coef[64 + c]  = b;
    coefh[c]      = (_Float16)a;
    coefh[64 + c] = (_Float16)b;
  }
}

// ---------------------------------------------------------------------------
// K3/K5/K6/K8: 64x(32*KSTEPS) GEMM via v_wmma_f32_16x16x32_f16.
//   Weight-stationary: all A tiles + bias live in registers for the whole
//   strip loop; __launch_bounds__(256,1) gives the VGPR budget (~210 worst
//   case for KSTEPS=4) so nothing spills to scratch.
//   BNRELU: packed-f16 per-channel affine + ReLU on the B operand.
//   OUTMODE 0: f16 [pos][64]   1: f32 reference layout [B,C,N,K]
//   OUTMODE 2: f16 [pos][64] + per-channel sum/sumsq accumulation (for ge BN)
// ---------------------------------------------------------------------------
template <int KSTEPS, bool BNRELU, int OUTMODE>
__global__ __launch_bounds__(256, 1)
void k_gemm(const _Float16* __restrict__ src0,
            const _Float16* __restrict__ src1,
            const _Float16* __restrict__ W,
            const float* __restrict__ bias,
            const _Float16* __restrict__ coefh,
            void* __restrict__ out,
            float* __restrict__ stats) {
  __shared__ float ssum[128];
  const int lane = threadIdx.x & 31;
  const int wave = threadIdx.x >> 5;
  const int sel  = lane >> 4;        // half-wave selector
  const int col  = lane & 15;        // position / A-row within tile
  if (OUTMODE == 2) {
    if (threadIdx.x < 128) ssum[threadIdx.x] = 0.f;
    __syncthreads();
  }
  const int wid = blockIdx.x * 8 + wave;
  const int NW  = gridDim.x * 8;

  // --- preload all A tiles (contiguous k = 16*sel + j order, same as B) into registers ---
  v16h areg[KSTEPS][4];
  #pragma unroll
  for (int ks = 0; ks < KSTEPS; ++ks)
    #pragma unroll
    for (int m = 0; m < 4; ++m) {
      const _Float16* wr = W + (size_t)(m*16 + col)*(KSTEPS*32) + ks*32;
      v8h alo = *(const v8h*)(wr + sel*16);
      v8h ahi = *(const v8h*)(wr + sel*16 + 8);
      #pragma unroll
      for (int j = 0; j < 8; ++j) { areg[ks][m][j] = alo[j]; areg[ks][m][8+j] = ahi[j]; }
    }
  // --- preload bias into the C-operand layout ---
  v8f bias8[4];
  #pragma unroll
  for (int m = 0; m < 4; ++m)
    #pragma unroll
    for (int r = 0; r < 8; ++r)
      bias8[m][r] = bias[m*16 + sel*8 + r];

  for (int s = wid; s < NSTRIP; s += NW) {
    if (s + NW < NSTRIP)             // prefetch next strip's activations
      __builtin_prefetch(src0 + ((size_t)(s + NW)*16 + col)*CDIM, 0, 1);

    v8f acc[4];
    #pragma unroll
    for (int m = 0; m < 4; ++m) acc[m] = bias8[m];

    #pragma unroll
    for (int ks = 0; ks < KSTEPS; ++ks) {
      const bool second = (KSTEPS == 4) && (ks >= 2);
      const _Float16* src = second ? src1 : src0;
      const int kof = second ? (ks - 2)*32 : ks*32;
      // B 32x16 (f16): lane holds 16 contiguous channels for position `col`
      v16h bv = *(const v16h*)(src + ((size_t)s*16 + col)*CDIM + kof + sel*16);
      if (BNRELU) {                  // packed-f16 BN + ReLU (KSTEPS==2 here)
        const int cb = ks*32 + sel*16;
        v16h ca = *(const v16h*)(coefh + cb);
        v16h cz = *(const v16h*)(coefh + 64 + cb);
        #pragma unroll
        for (int j = 0; j < 16; ++j) {
          _Float16 x = bv[j] * ca[j] + cz[j];
          bv[j] = (x > (_Float16)0) ? x : (_Float16)0;
        }
      }
      #pragma unroll
      for (int m = 0; m < 4; ++m)
        acc[m] = __builtin_amdgcn_wmma_f32_16x16x32_f16(
            false, areg[ks][m], false, bv, (short)0, acc[m], false, false);
    }

    if (OUTMODE == 0 || OUTMODE == 2) {
      _Float16* oh = (_Float16*)out;
      #pragma unroll
      for (int m = 0; m < 4; ++m) {
        v8h hv;
        #pragma unroll
        for (int r = 0; r < 8; ++r) hv[r] = (_Float16)acc[m][r];
        *(v8h*)(oh + ((size_t)s*16 + col)*CDIM + m*16 + sel*8) = hv;
      }
      if (OUTMODE == 2) {            // reduce over the 16 positions per channel
        #pragma unroll
        for (int m = 0; m < 4; ++m)
          #pragma unroll
          for (int r = 0; r < 8; ++r) {
            float v = acc[m][r], vq = v*v;
            for (int off = 1; off < 16; off <<= 1) {
              v  += __shfl_xor(v,  off, 32);
              vq += __shfl_xor(vq, off, 32);
            }
            if (col == 0) {
              const int c = m*16 + sel*8 + r;
              atomicAdd(&ssum[c], v);
              atomicAdd(&ssum[64 + c], vq);
            }
          }
      }
    } else {                         // f32 output, reference [B,C,N,K] layout
      float* of = (float*)out;
      const int b = s >> 12, n = s & (NDIM - 1);
      #pragma unroll
      for (int m = 0; m < 4; ++m)
        #pragma unroll
        for (int r = 0; r < 8; ++r) {
          const int c = m*16 + sel*8 + r;
          of[(((size_t)b*CDIM + c)*NDIM + n)*KDIM + col] = acc[m][r];
        }
    }
  }
  if (OUTMODE == 2) {
    __syncthreads();
    if (threadIdx.x < 128) atomicAdd(&stats[threadIdx.x], ssum[threadIdx.x]);
  }
}

// ---------------------------------------------------------------------------
// K4: per-point polar grid scatter -> normalize -> gather (in place on P_raw)
//     One wave per point n; R*A=24 bins x 64 channels lives in LDS.
//     LDS ops from one wave are in-order => no atomics needed.
// ---------------------------------------------------------------------------
__global__ __launch_bounds__(128, 1)
void k_polar(const unsigned* __restrict__ idxp,
             const float* __restrict__ wts,
             _Float16* __restrict__ pr) {
  __shared__ float sgrid[4][24*64 + 8];
  __shared__ float scnt[4][24];
  const int lane = threadIdx.x & 31;
  const int wave = threadIdx.x >> 5;
  float* G   = sgrid[wave];
  float* CNT = scnt[wave];
  const int wid = blockIdx.x * 4 + wave;

  for (int n = wid; n < NPTS; n += gridDim.x * 4) {
    for (int i = lane; i < 24*64; i += 32) G[i] = 0.f;
    for (int i = lane; i < 24;    i += 32) CNT[i] = 0.f;

    // stage this point's P_raw + bins + weights in registers
    float v0[16], v1[16], w4[16][4];
    unsigned pk[16];
    #pragma unroll
    for (int k = 0; k < 16; ++k) {
      const size_t p = (size_t)n*16 + k;
      v0[k] = (float)pr[p*CDIM + lane];
      v1[k] = (float)pr[p*CDIM + lane + 32];
      pk[k] = idxp[p];
      w4[k][0] = wts[p*4+0]; w4[k][1] = wts[p*4+1];
      w4[k][2] = wts[p*4+2]; w4[k][3] = wts[p*4+3];
    }
    asm volatile("s_wait_dscnt 0" ::: "memory");

    // bilinear scatter-add (serial over k within the wave => race-free)
    #pragma unroll
    for (int k = 0; k < 16; ++k) {
      const int ri0 = pk[k] & 1, ri1 = (pk[k] >> 1) & 1;
      const int ai0 = (pk[k] >> 2) & 15, ai1 = (pk[k] >> 6) & 15;
      const int bn[4] = { ri0*12 + ai0, ri0*12 + ai1, ri1*12 + ai0, ri1*12 + ai1 };
      #pragma unroll
      for (int q = 0; q < 4; ++q) {
        const float w = w4[k][q];
        G[bn[q]*64 + lane]      += w * v0[k];
        G[bn[q]*64 + lane + 32] += w * v1[k];
        if (lane == 0) CNT[bn[q]] += w;
      }
    }
    asm volatile("s_wait_dscnt 0" ::: "memory");

    // normalize by clipped weight counts
    for (int i = lane; i < 24*64; i += 32)
      G[i] /= fmaxf(CNT[i >> 6], EPS);
    asm volatile("s_wait_dscnt 0" ::: "memory");

    // gather at k=0's bins, blended with per-k weights; write P_aligned
    const int ri0 = pk[0] & 1, ri1 = (pk[0] >> 1) & 1;
    const int ai0 = (pk[0] >> 2) & 15, ai1 = (pk[0] >> 6) & 15;
    const int b00 = ri0*12 + ai0, b01 = ri0*12 + ai1;
    const int b10 = ri1*12 + ai0, b11 = ri1*12 + ai1;
    const float g00a = G[b00*64 + lane],      g01a = G[b01*64 + lane];
    const float g10a = G[b10*64 + lane],      g11a = G[b11*64 + lane];
    const float g00b = G[b00*64 + lane + 32], g01b = G[b01*64 + lane + 32];
    const float g10b = G[b10*64 + lane + 32], g11b = G[b11*64 + lane + 32];
    #pragma unroll
    for (int k = 0; k < 16; ++k) {
      const size_t p = (size_t)n*16 + k;
      const float oa = g00a*w4[k][0] + g01a*w4[k][1] + g10a*w4[k][2] + g11a*w4[k][3];
      const float ob = g00b*w4[k][0] + g01b*w4[k][1] + g10b*w4[k][2] + g11b*w4[k][3];
      pr[p*CDIM + lane]      = (_Float16)oa;
      pr[p*CDIM + lane + 32] = (_Float16)ob;
    }
    asm volatile("s_wait_dscnt 0" ::: "memory");
  }
}

// ---------------------------------------------------------------------------
extern "C" void kernel_launch(void* const* d_in, const int* in_sizes, int n_in,
                              void* d_out, int out_size, void* d_ws, size_t ws_size,
                              hipStream_t stream) {
  const float* NX    = (const float*)d_in[0];
  const float* CX    = (const float*)d_in[1];
  const float* CF_W1 = (const float*)d_in[2];
  const float* CF_B1 = (const float*)d_in[3];
  const float* CF_G  = (const float*)d_in[4];
  const float* CF_BE = (const float*)d_in[5];
  const float* CF_W2 = (const float*)d_in[6];
  const float* CF_B2 = (const float*)d_in[7];
  const float* PF_W1 = (const float*)d_in[8];
  const float* PF_B1 = (const float*)d_in[9];
  const float* PF_G  = (const float*)d_in[10];
  const float* PF_BE = (const float*)d_in[11];
  const float* PF_W2 = (const float*)d_in[12];
  const float* PF_B2 = (const float*)d_in[13];
  const float* AC_W  = (const float*)d_in[14];
  const float* AC_B  = (const float*)d_in[15];
  const float* AP_W  = (const float*)d_in[16];
  const float* AP_B  = (const float*)d_in[17];
  const float* GAMMA = (const float*)d_in[18];
  const float* GE_W1 = (const float*)d_in[19];
  const float* GE_B1 = (const float*)d_in[20];
  const float* GE_G  = (const float*)d_in[21];
  const float* GE_BE = (const float*)d_in[22];
  const float* GE_W2 = (const float*)d_in[23];
  const float* GE_B2 = (const float*)d_in[24];

  char* ws = (char*)d_ws;
  _Float16* HC    = (_Float16*)(ws + OFF_HC);     // -> reused as FUSED
  _Float16* HP    = (_Float16*)(ws + OFF_HP);     // -> reused as H_G
  _Float16* CRAW  = (_Float16*)(ws + OFF_CRAW);
  _Float16* PRAW  = (_Float16*)(ws + OFF_PRAW);   // -> P_aligned (in place)
  unsigned* IDX   = (unsigned*)(ws + OFF_IDX);
  float*    WTS   = (float*)(ws + OFF_WTS);
  _Float16* WC2   = (_Float16*)(ws + OFF_WC2);
  _Float16* WP2   = (_Float16*)(ws + OFF_WP2);
  _Float16* WG1   = (_Float16*)(ws + OFF_WG1);
  _Float16* WG2   = (_Float16*)(ws + OFF_WG2);
  _Float16* WFU   = (_Float16*)(ws + OFF_WFU);
  float*    BFU   = (float*)(ws + OFF_BFU);
  float*    STC   = (float*)(ws + OFF_STAT);
  float*    STP   = STC + 128;
  float*    STG   = STC + 256;
  float*    COEFC = (float*)(ws + OFF_COEF);
  float*    COEFP = COEFC + 128;
  float*    COEFG = COEFC + 256;
  _Float16* CHC   = (_Float16*)(ws + OFF_COEFH);
  _Float16* CHP   = CHC + 128;
  _Float16* CHG   = CHC + 256;
  _Float16* FUSED = HC;
  _Float16* HG    = HP;

  k_prep<<<1, 256, 0, stream>>>(CF_W2, PF_W2, GE_W1, GE_W2, AC_W, AP_W, AC_B, AP_B,
                                GAMMA, WC2, WP2, WG1, WG2, WFU, BFU, STC);
  k_geom<<<POS/256, 256, 0, stream>>>(NX, CX, CF_W1, CF_B1, PF_W1, PF_B1,
                                      HC, HP, IDX, WTS);
  k_stats<<<512, 256, 0, stream>>>(HC, STC);
  k_stats<<<512, 256, 0, stream>>>(HP, STP);
  k_finalize<<<1, 64, 0, stream>>>(STC, CF_G, CF_BE, COEFC, CHC);
  k_finalize<<<1, 64, 0, stream>>>(STP, PF_G, PF_BE, COEFP, CHP);

  // C_raw = cf_w2 x relu(bn(h_c)) + cf_b2 ; P_raw likewise
  k_gemm<2, true, 0><<<512, 256, 0, stream>>>(HC, nullptr, WC2, CF_B2, CHC,
                                              (void*)CRAW, nullptr);
  k_gemm<2, true, 0><<<512, 256, 0, stream>>>(HP, nullptr, WP2, PF_B2, CHP,
                                              (void*)PRAW, nullptr);
  // per-point polar scatter/normalize/gather (P_raw -> P_aligned in place)
  k_polar<<<256, 128, 0, stream>>>(IDX, WTS, PRAW);
  // fused = [ac_w | gamma*ap_w] x [C_raw ; P_aligned] + (ac_b + gamma*ap_b)
  k_gemm<4, false, 0><<<512, 256, 0, stream>>>(CRAW, PRAW, WFU, BFU, nullptr,
                                               (void*)FUSED, nullptr);
  // h_g = ge_w1 x fused + ge_b1, with BN-stat accumulation
  k_gemm<2, false, 2><<<512, 256, 0, stream>>>(FUSED, nullptr, WG1, GE_B1, nullptr,
                                               (void*)HG, STG);
  k_finalize<<<1, 64, 0, stream>>>(STG, GE_G, GE_BE, COEFG, CHG);
  // out = ge_w2 x relu(bn(h_g)) + ge_b2, f32, reference [B,C,N,K] layout
  k_gemm<2, true, 1><<<512, 256, 0, stream>>>(HG, nullptr, WG2, GE_B2, COEFG ? CHG : CHG,
                                              d_out, nullptr);
}
